// kmscm_38371237822844
// MI455X (gfx1250) — matrix operand order repaired and checked by reading, not compiled, not run on hardware
//
#include <hip/hip_runtime.h>

// ---------------------------------------------------------------------------
// MI455X (gfx1250, wave32) implementation.
// bf16 WMMA (v_wmma_f32_16x16x32_bf16) for both convolutions; fp32 everywhere
// else. Workload is HBM-bound (~350MB moved vs ~10 GFLOP of matrix work), so
// the convs are implicit-GEMMs staged through LDS, and the serial attention
// stages run in single-workgroup kernels with LDS-resident images.
// The erosion-reconstruction kernel stages its 64KB image with
// global_load_async_to_lds_b128 (ASYNCcnt path) since no conversion is needed.
// ---------------------------------------------------------------------------

typedef __attribute__((ext_vector_type(16))) __bf16 v16bf;
typedef __attribute__((ext_vector_type(8)))  float  v8f;

union FragAB { v16bf v; unsigned u[8]; };
union FragC  { v8f   v; float    f[8]; };

// ---- workspace layout (float offsets) -------------------------------------
static constexpr size_t OFF_XX    = 0;              // 8*32*128*128
static constexpr size_t OFF_XN    = 4194304;        // 32*16384 normalized rows
static constexpr size_t OFF_C     = 4718592;        // 8*16384 centroids
static constexpr size_t OFF_XSQ   = 4849664;        // 32
static constexpr size_t OFF_LAB   = 4849696;        // 32 (as int)
static constexpr size_t OFF_KINIT = 4849728;        // 16
static constexpr size_t OFF_V22   = 4849760;        // 8*16384
static constexpr size_t OFF_V22S  = 4980832;        // 8*16384
static constexpr size_t OFF_LAB2D = 5111904;        // 8*16384
static constexpr size_t OFF_DIL   = 5242976;        // 8*16384
static constexpr size_t OFF_FIN   = 5374048;        // 8*16384
static constexpr size_t OFF_ATT   = 5505120;        // 16384
static constexpr size_t OFF_Y     = 5521504;        // 8*64*16384

// ===========================================================================
// K1: conv down 3x3 stride2 pad1, C 64->32, implicit GEMM via WMMA bf16.
// Grid: 1024 blocks (b,oh). Block: 256 thr = 8 waves. Each wave: 16 output
// pixels (one M-tile) x both N-tiles (32 oc). K = kh*192 + kw*64 + ic.
// ===========================================================================
__global__ __launch_bounds__(256)
void k1_conv_down(const float* __restrict__ x, const float* __restrict__ wdn,
                  const float* __restrict__ bdn, float* __restrict__ xx)
{
  __shared__ __bf16 wlds[9 * 32 * 72];   // [kh*3+kw][oc][ic], ic padded to 72
  __shared__ __bf16 slab[256 * 72];      // [col][ic], ic padded to 72

  const int t  = threadIdx.x;
  const int b  = blockIdx.x >> 7;
  const int oh = blockIdx.x & 127;

  // stage weights as bf16, transposed to [tap][oc][ic]
  for (int f = t; f < 32 * 64 * 9; f += 256) {
    const int kw = f % 3, kh = (f / 3) % 3, ic = (f / 9) % 64, oc = f / 576;
    wlds[((kh * 3 + kw) * 32 + oc) * 72 + ic] = (__bf16)wdn[f];
  }

  const int wave = t >> 5, lane = t & 31;
  const int g = lane >> 4, ln = lane & 15;
  const int mbase = wave * 16;          // ow base of this wave's M-tile

  FragC c0, c1;
#pragma unroll
  for (int r = 0; r < 8; ++r) { c0.f[r] = 0.f; c1.f[r] = 0.f; }

  for (int kh = 0; kh < 3; ++kh) {
    const int irow = 2 * oh - 1 + kh;
    const bool rvalid = (irow >= 0) && (irow < 256);
    __syncthreads();  // weights staged / previous slab consumers done
    {
      const float* src = x + (((size_t)b * 64) * 256 + (size_t)(rvalid ? irow : 0)) * 256 + t;
      for (int ch = 0; ch < 64; ++ch) {
        const float v = rvalid ? src[(size_t)ch * 256 * 256] : 0.f;
        slab[t * 72 + ch] = (__bf16)v;
      }
      if (kh < 2) {  // prefetch next input row (global_prefetch_b8)
        const int nrow = 2 * oh + kh;
        if (nrow >= 0 && nrow < 256)
          __builtin_prefetch(x + (((size_t)b * 64) * 256 + nrow) * 256 + t, 0, 0);
      }
    }
    __syncthreads();

    for (int kw = 0; kw < 3; ++kw) {
      const int icol = 2 * (mbase + ln) - 1 + kw;  // input col for A-row ln
      const int colc = icol < 0 ? 0 : icol;
      const bool cv = icol >= 0;
      for (int half = 0; half < 2; ++half) {
        const int ics = half * 32;
        // A fragment (16x32 bf16): lane ln=row, group g selects K-half-octet
        FragAB a;
#pragma unroll
        for (int i = 0; i < 8; ++i) {
          const int k = (i < 4) ? (g * 8 + i * 2) : (16 + g * 8 + (i - 4) * 2);
          const unsigned val = *(const unsigned*)&slab[colc * 72 + ics + k];
          a.u[i] = cv ? val : 0u;
        }
        // B fragments (32x16 bf16): lane ln=col(oc), group g selects K-half-octet
        FragAB b0, b1;
#pragma unroll
        for (int i = 0; i < 8; ++i) {
          const int k = (i < 4) ? (g * 8 + i * 2) : (16 + g * 8 + (i - 4) * 2);
          b0.u[i] = *(const unsigned*)&wlds[((kh * 3 + kw) * 32 + ln) * 72 + ics + k];
          b1.u[i] = *(const unsigned*)&wlds[((kh * 3 + kw) * 32 + 16 + ln) * 72 + ics + k];
        }
        c0.v = __builtin_amdgcn_wmma_f32_16x16x32_bf16(false, a.v, false, b0.v,
                                                       (short)0, c0.v, false, false);
        c1.v = __builtin_amdgcn_wmma_f32_16x16x32_bf16(false, a.v, false, b1.v,
                                                       (short)0, c1.v, false, false);
      }
    }
  }

  // D layout: lane -> N=ln, VGPR r -> M = r + 8*g  (consecutive ow)
  const int ow0 = mbase + 8 * g;
#pragma unroll
  for (int ntile = 0; ntile < 2; ++ntile) {
    FragC& cc = ntile ? c1 : c0;
    const int oc = ntile * 16 + ln;
    const float bias = bdn[oc];
    float tmp[8];
#pragma unroll
    for (int r = 0; r < 8; ++r) tmp[r] = cc.f[r] + bias;
    float* dst = xx + (((size_t)b * 32 + oc) * 128 + oh) * 128 + ow0;
    *(float4*)(dst)     = *(float4*)&tmp[0];
    *(float4*)(dst + 4) = *(float4*)&tmp[4];
  }
}

// ===========================================================================
// K2a: per-row min/max normalize of x1s (32 rows x 16384), write Xn, xsq,
// and centroid init C = Xn[:8]. One block per row.
// x1s[i][j] = xx0[j%32][i*512 + j/32]
// ===========================================================================
__global__ __launch_bounds__(256)
void k2a_rownorm(const float* __restrict__ xx, float* __restrict__ Xn,
                 float* __restrict__ C, float* __restrict__ xsq)
{
  __shared__ float smn[256], smx[256], ssum[256];
  const int i = blockIdx.x;
  const int t = threadIdx.x;
  float mn = 1e30f, mx = -1e30f;
  for (int j = t; j < 16384; j += 256) {
    const float v = xx[(size_t)(j & 31) * 16384 + i * 512 + (j >> 5)];
    mn = fminf(mn, v); mx = fmaxf(mx, v);
  }
  smn[t] = mn; smx[t] = mx;
  __syncthreads();
  for (int s = 128; s > 0; s >>= 1) {
    if (t < s) { smn[t] = fminf(smn[t], smn[t + s]); smx[t] = fmaxf(smx[t], smx[t + s]); }
    __syncthreads();
  }
  mn = smn[0]; mx = smx[0];
  const float rng = (mx - mn == 0.f) ? 1.f : (mx - mn);
  float sq = 0.f;
  for (int j = t; j < 16384; j += 256) {
    const float v = xx[(size_t)(j & 31) * 16384 + i * 512 + (j >> 5)];
    const float xn = (v - mn) / rng;
    Xn[(size_t)i * 16384 + j] = xn;
    if (i < 8) C[(size_t)i * 16384 + j] = xn;
    sq += xn * xn;
  }
  ssum[t] = sq; __syncthreads();
  for (int s = 128; s > 0; s >>= 1) { if (t < s) ssum[t] += ssum[t + s]; __syncthreads(); }
  if (t == 0) xsq[i] = ssum[0];
}

// ===========================================================================
// K2c: k-means (32 points of dim 16384, k=8, 10 iters + final assign).
// Single 1024-thread workgroup; 32x8 dots reduced with LDS float atomics.
// ===========================================================================
__global__ __launch_bounds__(1024)
void k2c_kmeans(const float* __restrict__ Xn, float* __restrict__ C,
                const float* __restrict__ xsq, int* __restrict__ labels)
{
  __shared__ float sdot[32 * 8];
  __shared__ float scsq[8];
  __shared__ int   slab[32];
  __shared__ float scnt[8];
  const int t = threadIdx.x;
  const int irow = t >> 5;
  const int jofs = t & 31;

  for (int iter = 0; iter <= 10; ++iter) {
    if (t < 8) scsq[t] = 0.f;
    if (t < 256) sdot[t] = 0.f;
    __syncthreads();
    {  // ||C||^2 partials
      float p[8];
#pragma unroll
      for (int c = 0; c < 8; ++c) p[c] = 0.f;
      for (int j = t; j < 16384; j += 1024) {
#pragma unroll
        for (int c = 0; c < 8; ++c) { const float cv = C[(size_t)c * 16384 + j]; p[c] += cv * cv; }
      }
#pragma unroll
      for (int c = 0; c < 8; ++c) atomicAdd(&scsq[c], p[c]);
    }
    {  // X . C^T partials: thread owns row irow, stride-32 columns
      float acc[8];
#pragma unroll
      for (int c = 0; c < 8; ++c) acc[c] = 0.f;
      for (int j = jofs; j < 16384; j += 32) {
        const float xv = Xn[(size_t)irow * 16384 + j];
#pragma unroll
        for (int c = 0; c < 8; ++c) acc[c] += xv * C[(size_t)c * 16384 + j];
      }
#pragma unroll
      for (int c = 0; c < 8; ++c) atomicAdd(&sdot[irow * 8 + c], acc[c]);
    }
    __syncthreads();
    if (t < 32) {  // argmin (first min wins, as jnp.argmin)
      float best = 1e38f; int bl = 0;
#pragma unroll
      for (int c = 0; c < 8; ++c) {
        const float d = xsq[t] - 2.f * sdot[t * 8 + c] + scsq[c];
        if (d < best) { best = d; bl = c; }
      }
      slab[t] = bl;
    }
    __syncthreads();
    if (t == 0) {
#pragma unroll
      for (int c = 0; c < 8; ++c) scnt[c] = 0.f;
      for (int i2 = 0; i2 < 32; ++i2) scnt[slab[i2]] += 1.f;
    }
    __syncthreads();
    if (iter == 10) break;
    // centroid update (keep old centroid on empty cluster)
    for (int j = t; j < 16384; j += 1024) {
      float xv[32];
#pragma unroll
      for (int i2 = 0; i2 < 32; ++i2) xv[i2] = Xn[(size_t)i2 * 16384 + j];
#pragma unroll
      for (int c = 0; c < 8; ++c) {
        if (scnt[c] > 0.f) {
          float s = 0.f;
#pragma unroll
          for (int i2 = 0; i2 < 32; ++i2) s += (slab[i2] == c) ? xv[i2] : 0.f;
          C[(size_t)c * 16384 + j] = s / scnt[c];
        }
      }
    }
    __syncthreads();
  }
  if (t < 32) labels[t] = slab[t];
}

// ===========================================================================
// K2d: v22[c][p] = max over channels in cluster c of xx0[ch][p] (0 if empty)
// ===========================================================================
__global__ __launch_bounds__(256)
void k2d_v22(const float* __restrict__ xx, const int* __restrict__ labels,
             float* __restrict__ v22)
{
  __shared__ int slab[32];
  const int t = threadIdx.x;
  if (t < 32) slab[t] = labels[t];
  __syncthreads();
  const int p = blockIdx.x * 256 + t;
  float vmax[8]; int cnt[8];
#pragma unroll
  for (int c = 0; c < 8; ++c) { vmax[c] = -1e38f; cnt[c] = 0; }
  for (int ch = 0; ch < 32; ++ch) {
    const float v = xx[(size_t)ch * 16384 + p];
    const int c = slab[ch];
#pragma unroll
    for (int cc = 0; cc < 8; ++cc)
      if (cc == c) { vmax[cc] = fmaxf(vmax[cc], v); cnt[cc]++; }
  }
#pragma unroll
  for (int c = 0; c < 8; ++c)
    v22[(size_t)c * 16384 + p] = cnt[c] ? vmax[c] : 0.f;
}

// ===========================================================================
// K2e: scale_col per cluster row + 1D k-means linspace init (0 / 1)
// ===========================================================================
__global__ __launch_bounds__(256)
void k2e_scale(const float* __restrict__ v22, float* __restrict__ v22s,
               float* __restrict__ kinit)
{
  __shared__ float smn[256], smx[256];
  const int c = blockIdx.x, t = threadIdx.x;
  float mn = 1e30f, mx = -1e30f;
  for (int j = t; j < 16384; j += 256) {
    const float v = v22[(size_t)c * 16384 + j];
    mn = fminf(mn, v); mx = fmaxf(mx, v);
  }
  smn[t] = mn; smx[t] = mx; __syncthreads();
  for (int s = 128; s > 0; s >>= 1) {
    if (t < s) { smn[t] = fminf(smn[t], smn[t + s]); smx[t] = fmaxf(smx[t], smx[t + s]); }
    __syncthreads();
  }
  mn = smn[0]; mx = smx[0];
  const float rng = (mx - mn == 0.f) ? 1.f : (mx - mn);
  for (int j = t; j < 16384; j += 256)
    v22s[(size_t)c * 16384 + j] = (v22[(size_t)c * 16384 + j] - mn) / rng;
  if (t == 0) { kinit[c * 2 + 0] = 0.f; kinit[c * 2 + 1] = (mx > mn) ? 1.f : 0.f; }
}

// ===========================================================================
// K2f: 1-D 2-means, 10 iterations + final assign -> lab2d in {0,1}
// ===========================================================================
__global__ __launch_bounds__(256)
void k2f_kmeans1d(const float* __restrict__ v22s, const float* __restrict__ kinit,
                  float* __restrict__ lab2d)
{
  __shared__ float sC0, sC1, ss0, ss1, sn0, sn1;
  const int c = blockIdx.x, t = threadIdx.x;
  if (t == 0) { sC0 = kinit[c * 2]; sC1 = kinit[c * 2 + 1]; }
  __syncthreads();
  for (int iter = 0; iter < 10; ++iter) {
    if (t == 0) { ss0 = 0.f; ss1 = 0.f; sn0 = 0.f; sn1 = 0.f; }
    __syncthreads();
    const float C0 = sC0, C1 = sC1;
    float s0 = 0.f, s1 = 0.f, n0 = 0.f, n1 = 0.f;
    for (int j = t; j < 16384; j += 256) {
      const float v = v22s[(size_t)c * 16384 + j];
      const float d0 = (v - C0) * (v - C0), d1 = (v - C1) * (v - C1);
      if (d0 <= d1) { s0 += v; n0 += 1.f; } else { s1 += v; n1 += 1.f; }
    }
    atomicAdd(&ss0, s0); atomicAdd(&ss1, s1);
    atomicAdd(&sn0, n0); atomicAdd(&sn1, n1);
    __syncthreads();
    if (t == 0) {
      if (sn0 > 0.f) sC0 = ss0 / sn0;
      if (sn1 > 0.f) sC1 = ss1 / sn1;
    }
    __syncthreads();
  }
  const float C0 = sC0, C1 = sC1;
  for (int j = t; j < 16384; j += 256) {
    const float v = v22s[(size_t)c * 16384 + j];
    const float d0 = (v - C0) * (v - C0), d1 = (v - C1) * (v - C1);
    lab2d[(size_t)c * 16384 + j] = (d0 <= d1) ? 0.f : 1.f;
  }
}

// ===========================================================================
// K2g: 4x4 max dilation, padding ((2,1),(2,1)), -inf pad value
// ===========================================================================
__global__ __launch_bounds__(256)
void k2g_dilate(const float* __restrict__ lab2d, float* __restrict__ dil)
{
  const int idx = blockIdx.x * 256 + threadIdx.x;
  const int c = idx >> 14, p = idx & 16383;
  const int i = p >> 7, j = p & 127;
  float m = -1e38f;
  for (int di = -2; di <= 1; ++di) {
    const int ii = i + di; if (ii < 0 || ii > 127) continue;
    for (int dj = -2; dj <= 1; ++dj) {
      const int jj = j + dj; if (jj < 0 || jj > 127) continue;
      m = fmaxf(m, lab2d[(size_t)c * 16384 + ii * 128 + jj]);
    }
  }
  dil[idx] = m;
}

// ===========================================================================
// K2h: 100 iterations of r = max(erode_cross(r), mask), LDS-resident image.
// The 64KB seed image is staged memory->LDS with the CDNA5 async DMA path
// (global_load_async_to_lds_b128, ASYNCcnt), overlapped with the mask loads.
// Read-then-write phases with barriers avoid double buffering. Then
// fin = mod(lab2d - rec, 256).
// ===========================================================================
__global__ __launch_bounds__(1024)
void k2h_reconstruct(const float* __restrict__ dil, const float* __restrict__ lab2d,
                     float* __restrict__ fin)
{
  __shared__ float r[16384];   // 64KB of the 320KB WGP LDS
  const int c = blockIdx.x, t = threadIdx.x;

  // async DMA: dil[c] -> r[], 1024 threads x 4 x 16B = 64KB.
  // LDS aperture keeps the byte offset in the low 32 bits of the flat address.
  const float* src = dil + (size_t)c * 16384;
  const unsigned lds_base = (unsigned)(size_t)(&r[0]);
#pragma unroll
  for (int s = 0; s < 4; ++s) {
    const unsigned byteoff = (unsigned)(s * 16384 + t * 16);
    const unsigned ldsa = lds_base + byteoff;
    asm volatile("global_load_async_to_lds_b128 %0, %1, %2"
                 :: "v"(ldsa), "v"(byteoff), "s"(src) : "memory");
  }

  // mask loads overlap the async copy (LOADcnt path, independent of ASYNCcnt)
  float msk[16];
#pragma unroll
  for (int s = 0; s < 16; ++s)
    msk[s] = lab2d[(size_t)c * 16384 + s * 1024 + t];

  asm volatile("s_wait_asynccnt 0" ::: "memory");
  __syncthreads();

  for (int it = 0; it < 100; ++it) {
    float tmp[16];
#pragma unroll
    for (int s = 0; s < 16; ++s) {
      const int p = s * 1024 + t;
      const int i = p >> 7, j = p & 127;
      float e = r[p];
      if (i > 0)   e = fminf(e, r[p - 128]);
      if (i < 127) e = fminf(e, r[p + 128]);
      if (j > 0)   e = fminf(e, r[p - 1]);
      if (j < 127) e = fminf(e, r[p + 1]);
      tmp[s] = fmaxf(e, msk[s]);
    }
    __syncthreads();
#pragma unroll
    for (int s = 0; s < 16; ++s) r[s * 1024 + t] = tmp[s];
    __syncthreads();
  }
#pragma unroll
  for (int s = 0; s < 16; ++s) {
    const int p = s * 1024 + t;
    const float d = msk[s] - r[p];
    fin[(size_t)c * 16384 + p] = d - 256.f * floorf(d * (1.f / 256.f));
  }
}

// ===========================================================================
// K2i: att = mean over 8 maps
// ===========================================================================
__global__ __launch_bounds__(256)
void k2i_att(const float* __restrict__ fin, float* __restrict__ att)
{
  const int p = blockIdx.x * 256 + threadIdx.x;
  float s = 0.f;
#pragma unroll
  for (int c = 0; c < 8; ++c) s += fin[(size_t)c * 16384 + p];
  att[p] = s * 0.125f;
}

// ===========================================================================
// K3: fused (1+att)*xx -> 1x1 conv (32->64) -> +bias -> relu, WMMA bf16.
// (1+att) is per-pixel scalar => applied after the GEMM, bias unscaled.
// Grid: 512 blocks (b, 256-px chunk). Block 256 thr = 8 waves.
// ===========================================================================
__global__ __launch_bounds__(256)
void k3_conv_up(const float* __restrict__ xx, const float* __restrict__ wup,
                const float* __restrict__ bup, const float* __restrict__ att,
                float* __restrict__ y)
{
  __shared__ __bf16 aslab[256 * 36];   // [px][ic], padded
  __shared__ __bf16 wlds[64 * 36];     // [oc][ic], padded

  const int t  = threadIdx.x;
  const int b  = blockIdx.x >> 6;
  const int p0 = (blockIdx.x & 63) * 256;

  for (int f = t; f < 64 * 32; f += 256) {
    const int ic = f & 31, oc = f >> 5;
    wlds[oc * 36 + ic] = (__bf16)wup[f];
  }
  {
    const float* src = xx + ((size_t)b * 32) * 16384 + p0 + t;
    for (int ic = 0; ic < 32; ++ic)
      aslab[t * 36 + ic] = (__bf16)src[(size_t)ic * 16384];
  }
  __syncthreads();

  const int wave = t >> 5, lane = t & 31;
  const int g = lane >> 4, ln = lane & 15;

  FragAB bfr[4];
#pragma unroll
  for (int nt = 0; nt < 4; ++nt) {
#pragma unroll
    for (int i = 0; i < 8; ++i) {
      const int k = (i < 4) ? (g * 8 + i * 2) : (16 + g * 8 + (i - 4) * 2);
      bfr[nt].u[i] = *(const unsigned*)&wlds[(nt * 16 + ln) * 36 + k];
    }
  }

  for (int mi = 0; mi < 2; ++mi) {
    const int mt = wave * 2 + mi;
    FragAB a;
#pragma unroll
    for (int i = 0; i < 8; ++i) {
      const int k = (i < 4) ? (g * 8 + i * 2) : (16 + g * 8 + (i - 4) * 2);
      a.u[i] = *(const unsigned*)&aslab[(mt * 16 + ln) * 36 + k];
    }
#pragma unroll
    for (int nt = 0; nt < 4; ++nt) {
      FragC cfr;
#pragma unroll
      for (int r = 0; r < 8; ++r) cfr.f[r] = 0.f;
      cfr.v = __builtin_amdgcn_wmma_f32_16x16x32_bf16(false, a.v, false, bfr[nt].v,
                                                      (short)0, cfr.v, false, false);
      const int oc = nt * 16 + ln;
      const float bias = bup[oc];
      const int pl0 = mt * 16 + 8 * g;
      float tmp[8];
#pragma unroll
      for (int r = 0; r < 8; ++r) {
        const float s = 1.f + att[p0 + pl0 + r];
        tmp[r] = fmaxf(fmaf(s, cfr.f[r], bias), 0.f);
      }
      float* dst = y + ((size_t)b * 64 + oc) * 16384 + p0 + pl0;
      *(float4*)(dst)     = *(float4*)&tmp[0];
      *(float4*)(dst + 4) = *(float4*)&tmp[4];
    }
  }
}

// ===========================================================================
// K4: bilinear 128->256 (half-pixel, edge clamp) + residual add
// ===========================================================================
__global__ __launch_bounds__(256)
void k4_resize_add(const float* __restrict__ y, const float* __restrict__ x,
                   float* __restrict__ out)
{
  const size_t idx = (size_t)blockIdx.x * 256 + threadIdx.x;
  const int j = (int)(idx & 255);
  const int i = (int)((idx >> 8) & 255);
  const size_t bc = idx >> 16;
  const float sy = (i + 0.5f) * 0.5f - 0.5f;
  const float sx = (j + 0.5f) * 0.5f - 0.5f;
  const int i0 = (int)floorf(sy), j0 = (int)floorf(sx);
  const float fy = sy - (float)i0, fx = sx - (float)j0;
  const int i0c = i0 < 0 ? 0 : i0, i1c = (i0 + 1) > 127 ? 127 : (i0 + 1);
  const int j0c = j0 < 0 ? 0 : j0, j1c = (j0 + 1) > 127 ? 127 : (j0 + 1);
  const float* yb = y + bc * 16384;
  const float v00 = yb[i0c * 128 + j0c], v01 = yb[i0c * 128 + j1c];
  const float v10 = yb[i1c * 128 + j0c], v11 = yb[i1c * 128 + j1c];
  const float v0 = v00 + fx * (v01 - v00);
  const float v1 = v10 + fx * (v11 - v10);
  out[idx] = x[idx] + (v0 + fy * (v1 - v0));
}

// ===========================================================================
extern "C" void kernel_launch(void* const* d_in, const int* in_sizes, int n_in,
                              void* d_out, int out_size, void* d_ws, size_t ws_size,
                              hipStream_t stream)
{
  (void)in_sizes; (void)n_in; (void)out_size; (void)ws_size;
  const float* x   = (const float*)d_in[0];
  const float* wdn = (const float*)d_in[1];
  const float* bdn = (const float*)d_in[2];
  const float* wup = (const float*)d_in[3];
  const float* bup = (const float*)d_in[4];

  float* ws    = (float*)d_ws;
  float* xx    = ws + OFF_XX;
  float* Xn    = ws + OFF_XN;
  float* C     = ws + OFF_C;
  float* xsq   = ws + OFF_XSQ;
  int*   labels = (int*)(ws + OFF_LAB);
  float* kinit = ws + OFF_KINIT;
  float* v22   = ws + OFF_V22;
  float* v22s  = ws + OFF_V22S;
  float* lab2d = ws + OFF_LAB2D;
  float* dil   = ws + OFF_DIL;
  float* fin   = ws + OFF_FIN;
  float* att   = ws + OFF_ATT;
  float* y     = ws + OFF_Y;

  k1_conv_down  <<<1024,   256, 0, stream>>>(x, wdn, bdn, xx);
  k2a_rownorm   <<<32,     256, 0, stream>>>(xx, Xn, C, xsq);
  k2c_kmeans    <<<1,     1024, 0, stream>>>(Xn, C, xsq, labels);
  k2d_v22       <<<64,     256, 0, stream>>>(xx, labels, v22);
  k2e_scale     <<<8,      256, 0, stream>>>(v22, v22s, kinit);
  k2f_kmeans1d  <<<8,      256, 0, stream>>>(v22s, kinit, lab2d);
  k2g_dilate    <<<512,    256, 0, stream>>>(lab2d, dil);
  k2h_reconstruct<<<8,    1024, 0, stream>>>(dil, lab2d, fin);
  k2i_att       <<<64,     256, 0, stream>>>(fin, att);
  k3_conv_up    <<<512,    256, 0, stream>>>(xx, wup, bup, att, y);
  k4_resize_add <<<131072, 256, 0, stream>>>(y, x, (float*)d_out);
}
